// GNNModel_69535520522398
// MI455X (gfx1250) — compile-verified, matrix operand order checked by reading, not run on hardware
//
#include <hip/hip_runtime.h>

// ---------------------------------------------------------------- types
typedef __attribute__((ext_vector_type(16))) __bf16 v16bf;
typedef __attribute__((ext_vector_type(8)))  __bf16 v8bf;
typedef __attribute__((ext_vector_type(8)))  float  v8f;
typedef __attribute__((ext_vector_type(4)))  unsigned int v4u;
typedef __attribute__((ext_vector_type(4)))  int    v4i;
typedef __attribute__((ext_vector_type(8)))  int    v8i;

#define H 128

static __device__ __forceinline__ float sigmoidf_(float x) {
    return 1.0f / (1.0f + __expf(-x));
}

// ---------------------------------------------------------------- tiny precompute
// wa_va[k] = sum_j Wa[k,j]*va[j]   (and same for Ua) -> collapses attention GEMMs
__global__ void k_wava(const float* __restrict__ Wa, const float* __restrict__ Ua,
                       const float* __restrict__ va,
                       float* __restrict__ wava, float* __restrict__ uava) {
    int k = threadIdx.x;                       // 0..127
    float a = 0.f, b = 0.f;
    for (int j = 0; j < H; ++j) {
        float vv = va[j];
        a += Wa[k * H + j] * vv;
        b += Ua[k * H + j] * vv;
    }
    wava[k] = a;
    uava[k] = b;
}

// ---------------------------------------------------------------- node scores
__global__ void k_score(const int* __restrict__ x, const int* __restrict__ batch,
                        const int* __restrict__ userid,
                        const float* __restrict__ gitem, const float* __restrict__ uemb,
                        const float* __restrict__ wava, const float* __restrict__ uava,
                        float* __restrict__ score, int N) {
    int node = blockIdx.x * (blockDim.x >> 5) + (threadIdx.x >> 5);
    int lane = threadIdx.x & 31;
    if (node >= N) return;
    int xi = x[node] - 1;
    int u  = userid[batch[node]];
    float4 a = ((const float4*)(gitem + (size_t)xi * H))[lane];
    float4 b = ((const float4*)(uemb  + (size_t)u  * H))[lane];
    float4 wv = ((const float4*)wava)[lane];
    float4 uv = ((const float4*)uava)[lane];
    float s = a.x*wv.x + a.y*wv.y + a.z*wv.z + a.w*wv.w
            + b.x*uv.x + b.y*uv.y + b.z*uv.z + b.w*uv.w;
    for (int o = 16; o > 0; o >>= 1) s += __shfl_down(s, o, 32);
    if (lane == 0) score[node] = (s > 0.f) ? s : 0.01f * s;
}

// ---------------------------------------------------------------- global softmax stats
__global__ void k_stats(const float* __restrict__ score, float* __restrict__ red, int N) {
    __shared__ float sm[1024];
    int t = threadIdx.x;
    float m = -3.4e38f;
    for (int i = t; i < N; i += 1024) m = fmaxf(m, score[i]);
    sm[t] = m; __syncthreads();
    for (int o = 512; o > 0; o >>= 1) { if (t < o) sm[t] = fmaxf(sm[t], sm[t + o]); __syncthreads(); }
    float gmax = sm[0]; __syncthreads();
    float s = 0.f;
    for (int i = t; i < N; i += 1024) s += __expf(score[i] - gmax);
    sm[t] = s; __syncthreads();
    for (int o = 512; o > 0; o >>= 1) { if (t < o) sm[t] += sm[t + o]; __syncthreads(); }
    if (t == 0) { red[0] = gmax; red[1] = sm[0]; }
}

// ---------------------------------------------------------------- glob_node + hidden1 (fused)
__global__ void k_glob_hidden(const int* __restrict__ x, const float* __restrict__ item,
                              const float* __restrict__ gitem,
                              const float* __restrict__ score, const float* __restrict__ red,
                              float* __restrict__ glob, float* __restrict__ hidden1, int N) {
    int idx = blockIdx.x * blockDim.x + threadIdx.x;     // over N*32 float4s
    if (idx >= N * 32) return;
    int n = idx >> 5, q = idx & 31;
    int xi = x[n] - 1;
    float att = __expf(score[n] - red[0]) / red[1];
    float4 itf = ((const float4*)(gitem + (size_t)xi * H))[q];
    float4 ie  = ((const float4*)(item  + (size_t)xi * H))[q];
    float4 g = make_float4(att * itf.x, att * itf.y, att * itf.z, att * itf.w);
    ((float4*)(glob    + (size_t)n * H))[q] = g;
    ((float4*)(hidden1 + (size_t)n * H))[q] = make_float4(ie.x + g.x, ie.y + g.y, ie.z + g.z, ie.w + g.w);
}

// ---------------------------------------------------------------- f32 -> bf16 convert
__global__ void k_cvt(const float* __restrict__ s, __bf16* __restrict__ d, long n) {
    for (long i = blockIdx.x * (long)blockDim.x + threadIdx.x; i < n;
         i += (long)gridDim.x * blockDim.x)
        d[i] = (__bf16)s[i];
}

// ---------------------------------------------------------------- GCN degree / norm
__global__ void k_deg_init(float* __restrict__ deg, int N) {
    int n = blockIdx.x * blockDim.x + threadIdx.x;
    if (n < N) deg[n] = 1.0f;                      // self-loop weight
}
__global__ void k_deg_edge(const int* __restrict__ col, const float* __restrict__ ec,
                           const float* __restrict__ odi, float* __restrict__ deg, int E) {
    int e = blockIdx.x * blockDim.x + threadIdx.x;
    if (e < E) atomicAdd(deg + col[e], ec[e] * odi[e]);
}
__global__ void k_dinv(const float* __restrict__ deg, float* __restrict__ dinv, int N) {
    int n = blockIdx.x * blockDim.x + threadIdx.x;
    if (n < N) { float d = deg[n]; dinv[n] = (d > 0.f) ? rsqrtf(d) : 0.f; }
}

// ---------------------------------------------------------------- WMMA helpers
static __device__ __forceinline__ v16bf pack16(v8bf lo, v8bf hi) {
    v16bf r;
#pragma unroll
    for (int i = 0; i < 8; ++i) { r[i] = lo[i]; r[i + 8] = hi[i]; }
    return r;
}

// ---------------------------------------------------------------- generic bf16 WMMA GEMM
// C[M,Nc] = A[M,K] @ Bt[Nc,K]^T (+bias[Nc]); one wave per 16x16 tile, K step 32.
__global__ void k_gemm_bf16(const __bf16* __restrict__ A, const __bf16* __restrict__ Bt,
                            const float* __restrict__ bias, float* __restrict__ C,
                            int M, int Nc, int K) {
    int lane = threadIdx.x & 31;
    int wave = threadIdx.x >> 5;
    int ntn  = Nc >> 4;
    int ntm  = M  >> 4;
    int tile = blockIdx.x * (blockDim.x >> 5) + wave;
    if (tile >= ntm * ntn) return;
    int tm = tile / ntn, tn = tile % ntn;
    int m0 = tm << 4, n0 = tn << 4;
    int r  = lane & 15;
    int hh = (lane >> 4) << 3;                     // 0 or 8 halves
    const __bf16* arow = A  + (size_t)(m0 + r) * K + hh;
    const __bf16* brow = Bt + (size_t)(n0 + r) * K + hh;
    v8f c = {0.f, 0.f, 0.f, 0.f, 0.f, 0.f, 0.f, 0.f};
    for (int k0 = 0; k0 < K; k0 += 32) {
        if (k0 + 32 < K) {                         // global_prefetch_b8 (gfx1250)
            __builtin_prefetch(arow + k0 + 32, 0, 1);
            __builtin_prefetch(brow + k0 + 32, 0, 1);
        }
        v16bf av = pack16(*(const v8bf*)(arow + k0), *(const v8bf*)(arow + k0 + 16));
        v16bf bv = pack16(*(const v8bf*)(brow + k0), *(const v8bf*)(brow + k0 + 16));
        c = __builtin_amdgcn_wmma_f32_16x16x32_bf16(false, av, false, bv,
                                                    (short)0, c, false, false);
    }
    int colj = lane & 15;
    int rofs = (lane >> 4) << 3;
    float bs = bias ? bias[n0 + colj] : 0.0f;
#pragma unroll
    for (int g = 0; g < 8; ++g) {
        int m = m0 + g + rofs;
        C[(size_t)m * Nc + n0 + colj] = c[g] + bs;
    }
}

// ---------------------------------------------------------------- async-staged WMMA GEMM
// Block = one 16-row A tile (GLOBAL_LOAD_ASYNC_TO_LDS_B128 -> LDS, shared by 8
// waves) x strip of 8 column tiles. Launch: grid(ceil(Nc/128), M/16), 256 thr,
// dynamic LDS = 32*K bytes.
__global__ void k_gemm_bf16_lds(const __bf16* __restrict__ A, const __bf16* __restrict__ Bt,
                                const float* __restrict__ bias, float* __restrict__ C,
                                int M, int Nc, int K) {
    extern __shared__ __bf16 smemA[];              // 16 x K bf16
    const int lane = threadIdx.x & 31;
    const int wave = threadIdx.x >> 5;
    const int m0   = blockIdx.y << 4;
    const int ntn  = Nc >> 4;
    const int tn   = blockIdx.x * 8 + wave;

    {   // stage A tile: 32*K contiguous bytes, 16B per lane
        const __bf16* saddr = A + (size_t)m0 * K;          // uniform -> SGPR pair
        unsigned ldsBase = (unsigned)(size_t)(&smemA[0]);
        int nchunk = (K * 32) >> 4;                        // multiple of 256
        for (int ch = threadIdx.x; ch < nchunk; ch += blockDim.x) {
            unsigned ldsOff = ldsBase + (unsigned)(ch << 4);
            unsigned voff   = (unsigned)(ch << 4);
            asm volatile("global_load_async_to_lds_b128 %0, %1, %2"
                         :: "v"(ldsOff), "v"(voff), "s"(saddr)
                         : "memory");
        }
        asm volatile("s_wait_asynccnt 0x0" ::: "memory");
    }
    __syncthreads();

    if (tn < ntn) {                                         // wave-uniform guard
        const int r  = lane & 15;
        const int hh = (lane >> 4) << 3;
        const __bf16* arow = smemA + r * K + hh;            // LDS (ds_load_b128)
        const __bf16* brow = Bt + (size_t)((tn << 4) + r) * K + hh;
        v8f c = {0.f, 0.f, 0.f, 0.f, 0.f, 0.f, 0.f, 0.f};
        for (int k0 = 0; k0 < K; k0 += 32) {
            if (k0 + 32 < K)
                __builtin_prefetch(brow + k0 + 32, 0, 1);
            v16bf av = pack16(*(const v8bf*)(arow + k0), *(const v8bf*)(arow + k0 + 16));
            v16bf bv = pack16(*(const v8bf*)(brow + k0), *(const v8bf*)(brow + k0 + 16));
            c = __builtin_amdgcn_wmma_f32_16x16x32_bf16(false, av, false, bv,
                                                        (short)0, c, false, false);
        }
        const int n0   = tn << 4;
        const int colj = lane & 15;
        const int rofs = (lane >> 4) << 3;
        float bs = bias ? bias[n0 + colj] : 0.0f;
#pragma unroll
        for (int g = 0; g < 8; ++g) {
            int m = m0 + g + rofs;
            C[(size_t)m * Nc + n0 + colj] = c[g] + bs;
        }
    }
}

// ---------------------------------------------------------------- TDM-staged WMMA GEMM
// Block = 16-row A tile (async loads) x 128-col B strip moved by the Tensor
// Data Mover as ONE 2D descriptor (rows x K bf16, contiguous slab) into LDS.
// Waves then run pure ds_load_b128 -> v_wmma -> global_store. Sync:
// s_wait_asynccnt (per wave) + s_wait_tensorcnt (issuing wave) + barrier.
// Launch: grid(ceil(Nc/128), M/16), 256 thr, dynamic LDS = 32*K + 256*K bytes.
__global__ void k_gemm_bf16_tdm(const __bf16* __restrict__ A, const __bf16* __restrict__ Bt,
                                const float* __restrict__ bias, float* __restrict__ C,
                                int M, int Nc, int K) {
    extern __shared__ __bf16 smem[];
    __bf16* smemA = smem;                          // 16 x K
    __bf16* smemB = smem + 16 * K;                 // up to 128 x K
    const int lane   = threadIdx.x & 31;
    const int wave   = threadIdx.x >> 5;
    const int m0     = blockIdx.y << 4;
    const int ntn    = Nc >> 4;
    const int tn     = blockIdx.x * 8 + wave;
    const int strip0 = blockIdx.x * 128;           // first B row of this strip
    const int rows   = min(128, Nc - strip0);

    {   // A tile via async LDS loads (all 8 waves, 16B per lane)
        const __bf16* saddr = A + (size_t)m0 * K;
        unsigned ldsBase = (unsigned)(size_t)smemA;
        int nchunk = (K * 32) >> 4;
        for (int ch = threadIdx.x; ch < nchunk; ch += blockDim.x) {
            unsigned ldsOff = ldsBase + (unsigned)(ch << 4);
            unsigned voff   = (unsigned)(ch << 4);
            asm volatile("global_load_async_to_lds_b128 %0, %1, %2"
                         :: "v"(ldsOff), "v"(voff), "s"(saddr)
                         : "memory");
        }
    }
    if (wave == 0) {   // B strip via Tensor Data Mover (one D#, wave-level op)
        unsigned long long ga = (unsigned long long)(const void*)(Bt + (size_t)strip0 * K);
        unsigned ldsB = (unsigned)(size_t)smemB;
        unsigned td0 = (unsigned)K;                // elements per row
        unsigned td1 = (unsigned)rows;             // rows in this strip
        v4u g0; v8i g1; v4i g2 = {0, 0, 0, 0}, g3 = {0, 0, 0, 0};
        g0[0] = 1u;                                              // count=1 (valid user D#)
        g0[1] = ldsB;                                            // lds_addr
        g0[2] = (unsigned)ga;                                    // global_addr[31:0]
        g0[3] = (unsigned)((ga >> 32) & 0x01FFFFFFu) | (2u << 30); // addr[56:32] | type=2
        g1[0] = (int)(1u << 16);                                 // data_size=1 (2 bytes)
        g1[1] = (int)((td0 & 0xFFFFu) << 16);                    // tensor_dim0[15:0]
        g1[2] = (int)(((td0 >> 16) & 0xFFFFu) | ((td1 & 0xFFFFu) << 16)); // dim0 hi | dim1 lo
        g1[3] = (int)(((td1 >> 16) & 0xFFFFu) | ((td0 & 0xFFFFu) << 16)); // dim1 hi | tile_dim0=K
        g1[4] = (int)(td1 & 0xFFFFu);                            // tile_dim1=rows (tile_dim2=0)
        g1[5] = (int)td0;                                        // tensor_dim0_stride[31:0]=K
        g1[6] = 0;                                               // stride hi | dim1_stride lo
        g1[7] = 0;
#if __clang_major__ >= 23
        v8i g4 = {0, 0, 0, 0, 0, 0, 0, 0};
        __builtin_amdgcn_tensor_load_to_lds(g0, g1, g2, g3, g4, 0);
#else
        __builtin_amdgcn_tensor_load_to_lds(g0, g1, g2, g3, 0);
#endif
    }
    asm volatile("s_wait_asynccnt 0x0" ::: "memory");
    if (wave == 0) __builtin_amdgcn_s_wait_tensorcnt(0);
    __syncthreads();

    if (tn < ntn) {                                 // wave-uniform guard
        const int r  = lane & 15;
        const int hh = (lane >> 4) << 3;
        const __bf16* arow = smemA + r * K + hh;                 // LDS
        const __bf16* brow = smemB + (wave * 16 + r) * K + hh;   // LDS
        v8f c = {0.f, 0.f, 0.f, 0.f, 0.f, 0.f, 0.f, 0.f};
        for (int k0 = 0; k0 < K; k0 += 32) {
            v16bf av = pack16(*(const v8bf*)(arow + k0), *(const v8bf*)(arow + k0 + 16));
            v16bf bv = pack16(*(const v8bf*)(brow + k0), *(const v8bf*)(brow + k0 + 16));
            c = __builtin_amdgcn_wmma_f32_16x16x32_bf16(false, av, false, bv,
                                                        (short)0, c, false, false);
        }
        const int n0   = tn << 4;
        const int colj = lane & 15;
        const int rofs = (lane >> 4) << 3;
        float bs = bias ? bias[n0 + colj] : 0.0f;
#pragma unroll
        for (int g = 0; g < 8; ++g) {
            int m = m0 + g + rofs;
            C[(size_t)m * Nc + n0 + colj] = c[g] + bs;
        }
    }
}

// ---------------------------------------------------------------- GCN aggregate
__global__ void k_gcn_self(const float* __restrict__ xw, const float* __restrict__ dinv,
                           const float* __restrict__ bgcn, float* __restrict__ h2, int N) {
    int idx = blockIdx.x * blockDim.x + threadIdx.x;   // over N*32 float4s
    if (idx >= N * 32) return;
    int n = idx >> 5, q = idx & 31;
    float d2 = dinv[n] * dinv[n];
    float4 v = ((const float4*)(xw + (size_t)n * H))[q];
    float4 b = ((const float4*)bgcn)[q];
    ((float4*)(h2 + (size_t)n * H))[q] =
        make_float4(d2 * v.x + b.x, d2 * v.y + b.y, d2 * v.z + b.z, d2 * v.w + b.w);
}
__global__ void k_gcn_edge(const int* __restrict__ row, const int* __restrict__ col,
                           const float* __restrict__ ec, const float* __restrict__ odi,
                           const float* __restrict__ dinv, const float* __restrict__ xw,
                           float* __restrict__ h2, int E) {
    int idx = blockIdx.x * blockDim.x + threadIdx.x;   // over E*32 float4s
    if (idx >= E * 32) return;
    int e = idx >> 5, q = idx & 31;
    int rr = row[e], cc = col[e];
    float nrm = dinv[rr] * (ec[e] * odi[e]) * dinv[cc];
    float4 v = ((const float4*)(xw + (size_t)rr * H))[q];
    float* dst = h2 + (size_t)cc * H + q * 4;
    atomicAdd(dst + 0, nrm * v.x); atomicAdd(dst + 1, nrm * v.y);
    atomicAdd(dst + 2, nrm * v.z); atomicAdd(dst + 3, nrm * v.w);
}

// ---------------------------------------------------------------- last node per session
__global__ void k_lastidx(const int* __restrict__ batch, int* __restrict__ vidx, int N) {
    int n = blockIdx.x * blockDim.x + threadIdx.x;
    if (n >= N) return;
    if (n == N - 1 || batch[n + 1] != batch[n]) vidx[batch[n]] = n;
}

// ---------------------------------------------------------------- feat = [v_n_rep|h2|glob|u_n] in bf16
__global__ void k_feat(const float* __restrict__ h2, const float* __restrict__ glob,
                       const float* __restrict__ uemb,
                       const int* __restrict__ batch, const int* __restrict__ userid,
                       const int* __restrict__ vidx, __bf16* __restrict__ feat, int N) {
    int idx = blockIdx.x * blockDim.x + threadIdx.x;   // over N*128 groups of 4
    if (idx >= N * 128) return;
    int n = idx >> 7, g = idx & 127;
    int sec = g >> 5, q = g & 31;
    const float* src;
    if (sec == 0)      src = h2   + (size_t)vidx[batch[n]] * H;
    else if (sec == 1) src = h2   + (size_t)n * H;
    else if (sec == 2) src = glob + (size_t)n * H;
    else               src = uemb + (size_t)userid[batch[n]] * H;
    float4 v = ((const float4*)src)[q];
    __bf16* d = feat + (size_t)n * 512 + sec * H + q * 4;
    d[0] = (__bf16)v.x; d[1] = (__bf16)v.y; d[2] = (__bf16)v.z; d[3] = (__bf16)v.w;
}

// ---------------------------------------------------------------- alpha = sigmoid(pre) . W1 + b1
__global__ void k_alpha(const float* __restrict__ pre, const float* __restrict__ W1w,
                        const float* __restrict__ W1b, float* __restrict__ alpha, int N) {
    int node = blockIdx.x * (blockDim.x >> 5) + (threadIdx.x >> 5);
    int lane = threadIdx.x & 31;
    if (node >= N) return;
    float4 v = ((const float4*)(pre + (size_t)node * H))[lane];
    float4 w = ((const float4*)W1w)[lane];
    float s = sigmoidf_(v.x) * w.x + sigmoidf_(v.y) * w.y
            + sigmoidf_(v.z) * w.z + sigmoidf_(v.w) * w.w;
    for (int o = 16; o > 0; o >>= 1) s += __shfl_down(s, o, 32);
    if (lane == 0) alpha[node] = s + W1b[0];
}

// ---------------------------------------------------------------- s_g segment sum
__global__ void k_sg(const float* __restrict__ h2, const float* __restrict__ alpha,
                     const float* __restrict__ numc, const int* __restrict__ vidx,
                     float* __restrict__ sg, int B) {
    int idx = blockIdx.x * blockDim.x + threadIdx.x;   // over B*H
    if (idx >= B * H) return;
    int b = idx >> 7, q = idx & 127;
    int n0 = (b == 0) ? 0 : vidx[b - 1] + 1;
    int n1 = vidx[b];
    float s = 0.f;
    for (int n = n0; n <= n1; ++n)
        s += numc[n] * alpha[n] * h2[(size_t)n * H + q];
    sg[idx] = s;
}

// ---------------------------------------------------------------- [v_n | s_g] in bf16
__global__ void k_vnsg(const float* __restrict__ h2, const float* __restrict__ sg,
                       const int* __restrict__ vidx, __bf16* __restrict__ out, int B) {
    int idx = blockIdx.x * blockDim.x + threadIdx.x;   // over B*256
    if (idx >= B * 256) return;
    int b = idx >> 8, j = idx & 255;
    float v = (j < H) ? h2[(size_t)vidx[b] * H + j] : sg[(size_t)b * H + (j - H)];
    out[idx] = (__bf16)v;
}

// ---------------------------------------------------------------- launcher
static inline unsigned cdiv(long a, long b) { return (unsigned)((a + b - 1) / b); }

extern "C" void kernel_launch(void* const* d_in, const int* in_sizes, int n_in,
                              void* d_out, int out_size, void* d_ws, size_t ws_size,
                              hipStream_t stream) {
    const int*   x      = (const int*)d_in[0];
    const int*   eidx   = (const int*)d_in[1];
    const int*   batch  = (const int*)d_in[2];
    const int*   userid = (const int*)d_in[3];
    const float* numc   = (const float*)d_in[4];
    const float* ecnt   = (const float*)d_in[5];
    const float* odi    = (const float*)d_in[6];
    const float* item   = (const float*)d_in[7];
    const float* gitem  = (const float*)d_in[8];
    const float* uemb   = (const float*)d_in[9];
    const float* Wa     = (const float*)d_in[10];
    const float* Ua     = (const float*)d_in[11];
    const float* va     = (const float*)d_in[12];
    const float* Wgcn   = (const float*)d_in[13];
    const float* bgcn   = (const float*)d_in[14];
    const float* W1w    = (const float*)d_in[15];
    const float* W1b    = (const float*)d_in[16];
    const float* W2w    = (const float*)d_in[17];
    const float* W2b    = (const float*)d_in[18];
    const float* W5w    = (const float*)d_in[19];
    const float* W5b    = (const float*)d_in[20];

    const int N    = in_sizes[0];
    const int E    = in_sizes[5];
    const int B    = in_sizes[3];
    const int MAXI = out_size / B;

    // bump-allocate workspace
    char* p = (char*)d_ws;
    auto alloc = [&](size_t bytes) -> char* {
        char* r = p; p += (bytes + 255) & ~(size_t)255; return r;
    };
    float* wava   = (float*)alloc(H * 4);
    float* uava   = (float*)alloc(H * 4);
    float* score  = (float*)alloc((size_t)N * 4);
    float* red    = (float*)alloc(64 * 4);
    int*   vidx   = (int*)  alloc((size_t)B * 4);
    float* glob   = (float*)alloc((size_t)N * H * 4);
    float* hid1   = (float*)alloc((size_t)N * H * 4);   // reused as alpha_pre later
    float* deg    = (float*)alloc((size_t)N * 4);
    float* dinv   = (float*)alloc((size_t)N * 4);
    float* xw     = (float*)alloc((size_t)N * H * 4);
    float* hid2   = (float*)alloc((size_t)N * H * 4);
    float* alpha  = (float*)alloc((size_t)N * 4);
    float* sg     = (float*)alloc((size_t)B * H * 4);
    float* sh     = (float*)alloc((size_t)B * H * 4);
    __bf16* feat_bf = (__bf16*)alloc((size_t)N * 512 * 2);
    __bf16* hid1_bf = (__bf16*)alloc((size_t)N * H * 2);
    __bf16* wgcn_bf = (__bf16*)alloc((size_t)H * H * 2);
    __bf16* w2_bf   = (__bf16*)alloc((size_t)H * 512 * 2);
    __bf16* w5_bf   = (__bf16*)alloc((size_t)H * 256 * 2);
    __bf16* vnsg_bf = (__bf16*)alloc((size_t)B * 256 * 2);
    __bf16* sh_bf   = (__bf16*)alloc((size_t)B * H * 2);
    __bf16* item_bf = (__bf16*)alloc((size_t)MAXI * H * 2);

    const int* erow = eidx;
    const int* ecol = eidx + E;

    // 1) collapse attention weights
    k_wava<<<1, H, 0, stream>>>(Wa, Ua, va, wava, uava);
    // 2) per-node scores (wave per node)
    k_score<<<cdiv(N, 8), 256, 0, stream>>>(x, batch, userid, gitem, uemb, wava, uava, score, N);
    // 3) global softmax stats
    k_stats<<<1, 1024, 0, stream>>>(score, red, N);
    // 4) glob_node + hidden1
    k_glob_hidden<<<cdiv((long)N * 32, 256), 256, 0, stream>>>(x, item, gitem, score, red, glob, hid1, N);
    // 5) bf16 conversions for WMMA operands
    k_cvt<<<2048, 256, 0, stream>>>(hid1, hid1_bf, (long)N * H);
    k_cvt<<<64,   256, 0, stream>>>(Wgcn, wgcn_bf, (long)H * H);
    k_cvt<<<256,  256, 0, stream>>>(W2w,  w2_bf,   (long)H * 512);
    k_cvt<<<128,  256, 0, stream>>>(W5w,  w5_bf,   (long)H * 256);
    k_cvt<<<2048, 256, 0, stream>>>(item, item_bf, (long)MAXI * H);
    // 6) GCN normalization
    k_deg_init<<<cdiv(N, 256), 256, 0, stream>>>(deg, N);
    k_deg_edge<<<cdiv(E, 256), 256, 0, stream>>>(ecol, ecnt, odi, deg, E);
    k_dinv<<<cdiv(N, 256), 256, 0, stream>>>(deg, dinv, N);
    // 7) xw = hidden1 @ Wgcn^T   (async-LDS WMMA, N x 128 x 128)
    {
        dim3 grid(cdiv(H / 16, 8), N / 16);
        k_gemm_bf16_lds<<<grid, 256, 32 * H, stream>>>(hid1_bf, wgcn_bf, nullptr, xw, N, H, H);
    }
    // 8) hidden2 = dinv^2*xw + b  then scatter-add edges
    k_gcn_self<<<cdiv((long)N * 32, 256), 256, 0, stream>>>(xw, dinv, bgcn, hid2, N);
    k_gcn_edge<<<cdiv((long)E * 32, 256), 256, 0, stream>>>(erow, ecol, ecnt, odi, dinv, xw, hid2, E);
    // 9) last node of each session
    k_lastidx<<<cdiv(N, 256), 256, 0, stream>>>(batch, vidx, N);
    // 10) feat [N,512] bf16
    k_feat<<<cdiv((long)N * 128, 256), 256, 0, stream>>>(hid2, glob, uemb, batch, userid, vidx, feat_bf, N);
    // 11) alpha_pre = feat @ W2^T + b2   (async-LDS WMMA, N x 128 x 512, A reused 8x from LDS)
    {
        dim3 grid(cdiv(H / 16, 8), N / 16);
        k_gemm_bf16_lds<<<grid, 256, 32 * 512, stream>>>(feat_bf, w2_bf, W2b, hid1, N, H, 512);
    }
    // 12) alpha = sigmoid(alpha_pre) . W1 + b1
    k_alpha<<<cdiv(N, 8), 256, 0, stream>>>(hid1, W1w, W1b, alpha, N);
    // 13) s_g segment sum
    k_sg<<<cdiv((long)B * H, 256), 256, 0, stream>>>(hid2, alpha, numc, vidx, sg, B);
    // 14) [v_n | s_g] bf16
    k_vnsg<<<cdiv((long)B * 256, 256), 256, 0, stream>>>(hid2, sg, vidx, vnsg_bf, B);
    // 15) s_h = [v_n|s_g] @ W5^T + b5   (per-wave-tile WMMA, B x 128 x 256)
    {
        long tiles = (long)(B / 16) * (H / 16);
        k_gemm_bf16<<<cdiv(tiles, 8), 256, 0, stream>>>(vnsg_bf, w5_bf, W5b, sh, B, H, 256);
    }
    // 16) s_h -> bf16
    k_cvt<<<256, 256, 0, stream>>>(sh, sh_bf, (long)B * H);
    // 17) out = s_h @ item_emb[:MAXI]^T  (TDM-staged WMMA, 4096 x 20000 x 128)
    {
        dim3 grid(cdiv(MAXI / 16, 8), B / 16);
        k_gemm_bf16_tdm<<<grid, 256, (32 + 256) * H, stream>>>(sh_bf, item_bf, nullptr,
                                                               (float*)d_out, B, MAXI, H);
    }
    (void)n_in; (void)ws_size;
}
